// Kw2Poem_65386582114816
// MI455X (gfx1250) — matrix operand order repaired and checked by reading, not compiled, not run on hardware
//
#include <hip/hip_runtime.h>

// ---------------- problem constants ----------------
#define BB 128
#define LL 8
#define TT 48
#define VV 10000
#define EE 512
#define HH 512
#define FOURH 2048
#define NPAD_V 10112           // 10000 padded up to multiple of 128
#define M_ENC (BB*LL)          // 1024
#define M_DEC (BB*TT)          // 6144

// gfx1250 async global->LDS staging (ASYNCcnt-tracked). Flip to 0 to fall back
// to the register-staged global_load_b128 + ds_store_b128 path.
#define USE_ASYNC_LDS 1

typedef __attribute__((ext_vector_type(16))) __bf16 v16bf;
typedef __attribute__((ext_vector_type(8)))  float  v8f;

__device__ __forceinline__ float sigm(float x) { return 1.0f / (1.0f + __expf(-x)); }

#if USE_ASYNC_LDS
// per-lane async copy of 16B: LDS byte address (low 32 bits of generic pointer)
// in one VGPR, 64-bit global address in a VGPR pair, GV addressing mode.
#define ASYNC_COPY_B128(ldsdst, gsrc)                                          \
    asm volatile("global_load_async_to_lds_b128 %0, %1, off"                   \
                 :: "v"((unsigned)(unsigned long long)(const void*)(ldsdst)),  \
                    "v"((unsigned long long)(const void*)(gsrc))               \
                 : "memory")

__device__ __forceinline__ void async_wait0() {
#if __has_builtin(__builtin_amdgcn_s_wait_asynccnt)
    __builtin_amdgcn_s_wait_asynccnt(0);
#else
    asm volatile("s_wait_asynccnt 0x0" ::: "memory");
#endif
}
#endif

// ---------------- utility kernels ----------------
__global__ __launch_bounds__(256) void k_cvt_bf16(const float* __restrict__ src,
                                                  __bf16* __restrict__ dst, int n) {
    int i = blockIdx.x * 256 + threadIdx.x;
    if (i < n) dst[i] = (__bf16)src[i];
}

// out_W [10000,1024] f32 -> bf16 padded to [10112,1024] (extra rows zero)
__global__ __launch_bounds__(256) void k_cvt_pad_outw(const float* __restrict__ src,
                                                      __bf16* __restrict__ dst) {
    int i = blockIdx.x * 256 + threadIdx.x;
    int total = NPAD_V * (2 * HH);
    if (i >= total) return;
    int row = i >> 10;                       // /1024
    dst[i] = (row < VV) ? (__bf16)src[i] : (__bf16)0.0f;
}

__global__ __launch_bounds__(256) void k_gather_embed(const int* __restrict__ tok,
                                                      const float* __restrict__ emb,
                                                      __bf16* __restrict__ out, int total) {
    int i = blockIdx.x * 256 + threadIdx.x;
    if (i >= total) return;
    int row = i >> 9;                        // /512
    int col = i & 511;
    out[i] = (__bf16)emb[(size_t)tok[row] * EE + col];
}

__global__ __launch_bounds__(256) void k_init_state(float* __restrict__ h,
                                                    float* __restrict__ c,
                                                    __bf16* __restrict__ hb) {
    int i = blockIdx.x * 256 + threadIdx.x;  // 65536 threads
    h[i] = 0.0f; c[i] = 0.0f; hb[i] = (__bf16)0.0f;
}

// ---------------- generic bf16 WMMA GEMM: C = A * B^T (+bias)(+add) ----------------
// A: [M,K] bf16 row-major (lda), B: [Npad,K] bf16 row-major (ldb, i.e. B^T is KxN)
// C: [M,Nstore] f32 (ldc). Block tile 128x128, 8 waves of 32x64, K-step 32.
// LDS is double-buffered; tiles staged with async global->LDS copies so the
// next tile's fetch overlaps the current tile's WMMAs (one barrier per K-step).
__global__ __launch_bounds__(256) void k_gemm_bf16(
    const __bf16* __restrict__ A, int lda,
    const __bf16* __restrict__ Bw, int ldb,
    float* __restrict__ C, int ldc,
    const float* __restrict__ bias,
    const float* __restrict__ addmat, int add_ld,
    int Nstore, int K)
{
    __shared__ alignas(16) __bf16 As[2][128][40];  // 32 K + 8 pad (80B rows, 16B aligned)
    __shared__ alignas(16) __bf16 Bs[2][128][40];

    const int tid  = threadIdx.x;
    const int m0   = blockIdx.y * 128;
    const int n0   = blockIdx.x * 128;
    const int wave = tid >> 5;
    const int lane = tid & 31;
    const int wm   = (wave & 3) * 32;            // wave M offset in tile
    const int wn   = (wave >> 2) * 64;           // wave N offset in tile
    const int r16  = lane & 15;
    const int half = lane >> 4;

    v8f acc[2][4];
    #pragma unroll
    for (int i = 0; i < 2; ++i)
        #pragma unroll
        for (int j = 0; j < 4; ++j)
            #pragma unroll
            for (int e = 0; e < 8; ++e) acc[i][j][e] = 0.0f;

    auto stage = [&](int buf, int k0) {
        #pragma unroll
        for (int i = 0; i < 2; ++i) {
            int chunk = tid + i * 256;           // 0..511
            int row   = chunk >> 2;
            int kc    = (chunk & 3) << 3;        // 0,8,16,24
            const __bf16* ga = A  + (size_t)(m0 + row) * lda + k0 + kc;
            const __bf16* gb = Bw + (size_t)(n0 + row) * ldb + k0 + kc;
#if USE_ASYNC_LDS
            ASYNC_COPY_B128(&As[buf][row][kc], ga);
            ASYNC_COPY_B128(&Bs[buf][row][kc], gb);
#else
            *(int4*)(&As[buf][row][kc]) = *(const int4*)ga;
            *(int4*)(&Bs[buf][row][kc]) = *(const int4*)gb;
#endif
        }
    };

    stage(0, 0);
    int buf = 0;
    for (int k0 = 0; k0 < K; k0 += 32) {
#if USE_ASYNC_LDS
        async_wait0();                           // this wave's copies landed in LDS
#endif
        __syncthreads();                         // all waves' copies landed; prior reads done
        if (k0 + 32 < K) stage(buf ^ 1, k0 + 32);

        // A fragments: lane-half holds K chunks {half*8 .. +7} and {16+half*8 .. +7}
        v16bf af[2];
        #pragma unroll
        for (int am = 0; am < 2; ++am) {
            int row = wm + am * 16 + r16;
            const __bf16* p0 = &As[buf][row][half * 8];
            const __bf16* p1 = &As[buf][row][16 + half * 8];
            #pragma unroll
            for (int e = 0; e < 8; ++e) { af[am][e] = p0[e]; af[am][8 + e] = p1[e]; }
        }
        // B fragments: lane-half holds K chunks {half*8 .. +7} and {16+half*8 .. +7} of column n
        v16bf bfr[4];
        #pragma unroll
        for (int bn = 0; bn < 4; ++bn) {
            int row = wn + bn * 16 + r16;        // row of W == column of B^T
            const __bf16* p0 = &Bs[buf][row][half * 8];
            const __bf16* p1 = &Bs[buf][row][16 + half * 8];
            #pragma unroll
            for (int e = 0; e < 8; ++e) { bfr[bn][e] = p0[e]; bfr[bn][8 + e] = p1[e]; }
        }

        #pragma unroll
        for (int am = 0; am < 2; ++am)
            #pragma unroll
            for (int bn = 0; bn < 4; ++bn)
                acc[am][bn] = __builtin_amdgcn_wmma_f32_16x16x32_bf16(
                    false, af[am], false, bfr[bn], (short)0, acc[am][bn], false, false);

        buf ^= 1;
    }

    // store: C/D layout -> lane col = r16 (+16 lanes => rows +8), VGPR r => row r
    #pragma unroll
    for (int am = 0; am < 2; ++am) {
        #pragma unroll
        for (int bn = 0; bn < 4; ++bn) {
            int gcol = n0 + wn + bn * 16 + r16;
            if (gcol < Nstore) {
                float bv = bias ? bias[gcol] : 0.0f;
                #pragma unroll
                for (int r = 0; r < 8; ++r) {
                    int gm = m0 + wm + am * 16 + half * 8 + r;
                    float v = acc[am][bn][r] + bv;
                    if (addmat) v += addmat[(size_t)gm * add_ld + gcol];
                    C[(size_t)gm * ldc + gcol] = v;
                }
            }
        }
    }
}

// ---------------- encoder step activation ----------------
__global__ __launch_bounds__(256) void k_enc_act(
    const float* __restrict__ gates,   // [128,2048] = x_proj(+b) + h@Whh^T
    float* __restrict__ h_state, float* __restrict__ c_state,
    __bf16* __restrict__ h_bf,
    float* __restrict__ hs, float* __restrict__ cs, int l)
{
    int idx = blockIdx.x * 256 + threadIdx.x;   // 65536 = B*H
    int b = idx >> 9, j = idx & 511;
    const float* g = gates + (size_t)b * FOURH;
    float ig = g[j], fg = g[512 + j], gg = g[1024 + j], og = g[1536 + j];
    float c = sigm(fg) * c_state[idx] + sigm(ig) * tanhf(gg);
    float h = sigm(og) * tanhf(c);
    c_state[idx] = c; h_state[idx] = h; h_bf[idx] = (__bf16)h;
    hs[(size_t)l * (BB * HH) + idx] = h;
    cs[(size_t)l * (BB * HH) + idx] = c;
}

// select (h0,c0) at last valid timestep; build masked kws_rep [B,L,H]
__global__ __launch_bounds__(256) void k_enc_finalize(
    const float* __restrict__ hs, const float* __restrict__ cs,
    const int* __restrict__ kw_lens,
    float* __restrict__ h_state, float* __restrict__ c_state,
    __bf16* __restrict__ h_bf, float* __restrict__ kws_rep)
{
    int idx = blockIdx.x * 256 + threadIdx.x;   // B*H
    int b = idx >> 9, j = idx & 511;
    int len = kw_lens[b];
    float h = hs[(size_t)(len - 1) * (BB * HH) + idx];
    float c = cs[(size_t)(len - 1) * (BB * HH) + idx];
    h_state[idx] = h; c_state[idx] = c; h_bf[idx] = (__bf16)h;
    #pragma unroll
    for (int l = 0; l < LL; ++l)
        kws_rep[(size_t)b * (LL * HH) + l * HH + j] =
            (l < len) ? hs[(size_t)l * (BB * HH) + idx] : 0.0f;
}

// ---------------- decoder step: LSTM act + attention + pack [h|ctx] ----------------
__global__ __launch_bounds__(256) void k_dec_act_attn(
    const float* __restrict__ gates,     // [128,2048]
    float* __restrict__ c_state,         // [128,512]
    __bf16* __restrict__ h_bf,           // [128,512] (next-step GEMM input)
    const float* __restrict__ kws_rep,   // [128,8,512]
    const int* __restrict__ kw_lens,
    __bf16* __restrict__ hc,             // [B*T,1024], row = b*T+t
    int t)
{
    __shared__ float hsm[HH];
    __shared__ float red[256];
    __shared__ float attn[LL];
    int b = blockIdx.x, tid = threadIdx.x;
    const float* g = gates + (size_t)b * FOURH;

    float hv[2];
    #pragma unroll
    for (int q = 0; q < 2; ++q) {
        int j = tid + q * 256;
        float ig = g[j], fg = g[512 + j], gg = g[1024 + j], og = g[1536 + j];
        float c = sigm(fg) * c_state[b * HH + j] + sigm(ig) * tanhf(gg);
        float h = sigm(og) * tanhf(c);
        c_state[b * HH + j] = c;
        hsm[j] = h;
        hv[q] = h;
    }
    __syncthreads();

    const float* kr = kws_rep + (size_t)b * (LL * HH);
    for (int l = 0; l < LL; ++l) {
        float p = hv[0] * kr[l * HH + tid] + hv[1] * kr[l * HH + tid + 256];
        red[tid] = p; __syncthreads();
        for (int s = 128; s > 0; s >>= 1) {
            if (tid < s) red[tid] += red[tid + s];
            __syncthreads();
        }
        if (tid == 0) attn[l] = red[0];
        __syncthreads();
    }
    if (tid == 0) {
        int len = kw_lens[b];
        float mx = -1e30f;
        for (int l = 0; l < len; ++l) mx = fmaxf(mx, attn[l]);
        float sum = 0.0f;
        for (int l = 0; l < LL; ++l) {
            float e = (l < len) ? __expf(attn[l] - mx) : 0.0f;
            attn[l] = e; sum += e;
        }
        float inv = 1.0f / sum;
        for (int l = 0; l < LL; ++l) attn[l] *= inv;
    }
    __syncthreads();

    size_t row = (size_t)(b * TT + t) * (2 * HH);
    #pragma unroll
    for (int q = 0; q < 2; ++q) {
        int j = tid + q * 256;
        float ctx = 0.0f;
        #pragma unroll
        for (int l = 0; l < LL; ++l) ctx += attn[l] * kr[l * HH + j];
        hc[row + j]       = (__bf16)hsm[j];
        hc[row + HH + j]  = (__bf16)ctx;
        h_bf[b * HH + j]  = (__bf16)hsm[j];
    }
}

// ---------------- host orchestration ----------------
extern "C" void kernel_launch(void* const* d_in, const int* in_sizes, int n_in,
                              void* d_out, int out_size, void* d_ws, size_t ws_size,
                              hipStream_t stream) {
    const int*   kws    = (const int*)d_in[0];
    const int*   poem   = (const int*)d_in[1];
    const int*   kwlen  = (const int*)d_in[2];
    const float* emb    = (const float*)d_in[3];
    const float* eWih   = (const float*)d_in[4];
    const float* eWhh   = (const float*)d_in[5];
    const float* eb     = (const float*)d_in[6];
    const float* dWih   = (const float*)d_in[7];
    const float* dWhh   = (const float*)d_in[8];
    const float* db     = (const float*)d_in[9];
    const float* oW     = (const float*)d_in[10];
    const float* ob     = (const float*)d_in[11];
    float* out = (float*)d_out;

    // workspace carve-up (256B aligned regions)
    char* base = (char*)d_ws;
    size_t off = 0;
    auto alloc = [&](size_t bytes) -> char* {
        off = (off + 255) & ~(size_t)255;
        char* p = base + off;
        off += bytes;
        return p;
    };
    __bf16* eWih_bf  = (__bf16*)alloc((size_t)FOURH * EE * 2);
    __bf16* eWhh_bf  = (__bf16*)alloc((size_t)FOURH * HH * 2);
    __bf16* dWih_bf  = (__bf16*)alloc((size_t)FOURH * EE * 2);
    __bf16* dWhh_bf  = (__bf16*)alloc((size_t)FOURH * HH * 2);
    __bf16* oW_bf    = (__bf16*)alloc((size_t)NPAD_V * 2 * HH * 2);
    __bf16* kwsEmb   = (__bf16*)alloc((size_t)M_ENC * EE * 2);
    __bf16* poemEmb  = (__bf16*)alloc((size_t)M_DEC * EE * 2);
    float*  encX     = (float*)alloc((size_t)M_ENC * FOURH * 4);
    float*  decX     = (float*)alloc((size_t)M_DEC * FOURH * 4);
    float*  gates    = (float*)alloc((size_t)BB * FOURH * 4);
    float*  hs       = (float*)alloc((size_t)LL * BB * HH * 4);
    float*  cs       = (float*)alloc((size_t)LL * BB * HH * 4);
    float*  kwsRep   = (float*)alloc((size_t)BB * LL * HH * 4);
    float*  h_state  = (float*)alloc((size_t)BB * HH * 4);
    float*  c_state  = (float*)alloc((size_t)BB * HH * 4);
    __bf16* h_bf     = (__bf16*)alloc((size_t)BB * HH * 2);
    __bf16* hc       = (__bf16*)alloc((size_t)M_DEC * 2 * HH * 2);

    auto cdiv = [](int a, int b) { return (a + b - 1) / b; };

    // 0) init recurrent state
    k_init_state<<<(BB * HH) / 256, 256, 0, stream>>>(h_state, c_state, h_bf);

    // 1) weight conversions to bf16 (out_W zero-padded to NPAD_V rows)
    int nw = FOURH * EE;
    k_cvt_bf16<<<cdiv(nw, 256), 256, 0, stream>>>(eWih, eWih_bf, nw);
    k_cvt_bf16<<<cdiv(nw, 256), 256, 0, stream>>>(eWhh, eWhh_bf, nw);
    k_cvt_bf16<<<cdiv(nw, 256), 256, 0, stream>>>(dWih, dWih_bf, nw);
    k_cvt_bf16<<<cdiv(nw, 256), 256, 0, stream>>>(dWhh, dWhh_bf, nw);
    k_cvt_pad_outw<<<cdiv(NPAD_V * 2 * HH, 256), 256, 0, stream>>>(oW, oW_bf);

    // 2) embedding gathers (bf16)
    k_gather_embed<<<cdiv(M_ENC * EE, 256), 256, 0, stream>>>(kws, emb, kwsEmb, M_ENC * EE);
    k_gather_embed<<<cdiv(M_DEC * EE, 256), 256, 0, stream>>>(poem, emb, poemEmb, M_DEC * EE);

    // 3) batched input projections: X @ Wih^T + b  (all timesteps at once)
    k_gemm_bf16<<<dim3(FOURH / 128, M_ENC / 128), 256, 0, stream>>>(
        kwsEmb, EE, eWih_bf, EE, encX, FOURH, eb, nullptr, 0, FOURH, EE);
    k_gemm_bf16<<<dim3(FOURH / 128, M_DEC / 128), 256, 0, stream>>>(
        poemEmb, EE, dWih_bf, EE, decX, FOURH, db, nullptr, 0, FOURH, EE);

    // 4) encoder recurrence: gates = encX[:,l] + h@Whh^T ; LSTM activations
    for (int l = 0; l < LL; ++l) {
        k_gemm_bf16<<<dim3(FOURH / 128, 1), 256, 0, stream>>>(
            h_bf, HH, eWhh_bf, HH, gates, FOURH, nullptr,
            encX + (size_t)l * FOURH, LL * FOURH, FOURH, HH);
        k_enc_act<<<(BB * HH) / 256, 256, 0, stream>>>(
            gates, h_state, c_state, h_bf, hs, cs, l);
    }
    k_enc_finalize<<<(BB * HH) / 256, 256, 0, stream>>>(
        hs, cs, kwlen, h_state, c_state, h_bf, kwsRep);

    // 5) decoder recurrence: gates GEMM + fused activation/attention
    for (int t = 0; t < TT; ++t) {
        k_gemm_bf16<<<dim3(FOURH / 128, 1), 256, 0, stream>>>(
            h_bf, HH, dWhh_bf, HH, gates, FOURH, nullptr,
            decX + (size_t)t * FOURH, TT * FOURH, FOURH, HH);
        k_dec_act_attn<<<BB, 256, 0, stream>>>(
            gates, c_state, h_bf, kwsRep, kwlen, hc, t);
    }

    // 6) batched output projection: [B*T,1024] @ out_W^T + b -> d_out [B,T,V]
    k_gemm_bf16<<<dim3(NPAD_V / 128, M_DEC / 128), 256, 0, stream>>>(
        hc, 2 * HH, oW_bf, 2 * HH, out, VV, ob, nullptr, 0, VV, 2 * HH);
}
